// RoleAwareGraphTransformer_65859028517070
// MI455X (gfx1250) — matrix operand order repaired and checked by reading, not compiled, not run on hardware
//
#include <hip/hip_runtime.h>
#include <hip/hip_bf16.h>

#define N_STOCKS 20000
#define IN_F     64
#define PE_F     32
#define TIN      96
#define HIDF     128
#define HEADS    4
#define CDIM     32
#define E_EDGES  100000
#define R_REL    5
#define EAUG     (E_EDGES + N_STOCKS)   // edges + self loops
#define LDS_STRIDE 136                  // 128 + 8 halves pad (68 dwords -> min bank conflicts)

typedef __attribute__((ext_vector_type(16))) _Float16 v16h;
typedef __attribute__((ext_vector_type(8)))  float    v8f;

union AFrag { unsigned int u[8]; v16h v; };
union BFrag { uint4 q[2];        v16h v; };

// ---------------- f32 -> f16 weight conversion ----------------
__global__ __launch_bounds__(256)
void cvt_f16_kernel(const float* __restrict__ src, _Float16* __restrict__ dst, int count) {
    int t = blockIdx.x * 256 + threadIdx.x;
    if (t < count) dst[t] = (_Float16)src[t];
}

// ---------------- pearl PE + concat -> h16 [N,128] (f16) ----------------
__global__ __launch_bounds__(32)
void pearl_concat_kernel(const float* __restrict__ x, const float* __restrict__ pW,
                         const float* __restrict__ pb, _Float16* __restrict__ h16) {
    int row = blockIdx.x;
    int t   = threadIdx.x;               // 0..31
    const float* xr = x + (size_t)row * IN_F;
    float acc = pb[t];
#pragma unroll 8
    for (int k = 0; k < IN_F; ++k) acc += xr[k] * pW[k * PE_F + t];
    _Float16* hr = h16 + (size_t)row * HIDF;
    hr[t]      = (_Float16)xr[t];
    hr[t + 32] = (_Float16)xr[t + 32];
    hr[t + 64] = (_Float16)acc;
    hr[t + 96] = (_Float16)0.f;          // pad (layer0 only reads K=96)
}

// ---------------- fused dual WMMA GEMM with LDS-staged weights ----------------
// YL = X @ WL + biasL ; YR = X @ WR + biasR  (A fragment reused for both).
// Block = 4 waves. WL|WR staged once per block into LDS (padded rows), B fragments
// then come from 16-bit LDS reads strided by LDS_STRIDE (short-latency waits) while A streams from global.
__global__ __launch_bounds__(128)
void wmma_dual_lin_kernel(const _Float16* __restrict__ X, int K,
                          const _Float16* __restrict__ WL,
                          const _Float16* __restrict__ WR,
                          const float* __restrict__ biasL,
                          const float* __restrict__ biasR,
                          float* __restrict__ YL, float* __restrict__ YR,
                          int nrows) {
    extern __shared__ _Float16 ldsW[];   // [2][K][LDS_STRIDE]
    int tid  = threadIdx.x;
    int wave = tid >> 5;
    int lane = tid & 31;

    // ---- cooperative stage WL then WR into LDS, b128 chunks ----
    {
        int chunksPerMat = K * 16;                 // 16 chunks of 8 halves per row
        int total = 2 * chunksPerMat;
        for (int ch = tid; ch < total; ch += 128) {
            int m   = (ch >= chunksPerMat) ? 1 : 0;
            int rem = ch - m * chunksPerMat;
            int k   = rem >> 4;
            int c   = (rem & 15) * 8;
            const _Float16* src = (m ? WR : WL) + (size_t)k * HIDF + c;
            _Float16* dst = ldsW + (size_t)(m * K + k) * LDS_STRIDE + c;
            *(uint4*)dst = *(const uint4*)src;
        }
    }
    __syncthreads();

    int row0 = (blockIdx.x * 4 + wave) * 16;
    if (row0 >= nrows) return;

    v8f accL[8] = {};
    v8f accR[8] = {};
    int g = lane >> 4;                             // lane group
    const _Float16* arow = X + (size_t)(row0 + (lane & 15)) * HIDF;
    for (int k0 = 0; k0 < K; k0 += 32) {
        // prefetch next K-step of the A stream (global_prefetch_b8)
        if (k0 + 32 < K) __builtin_prefetch(arow + k0 + 32, 0, 1);
        // A fragment: 16x32 f16, lane holds row M=lane%16; K mapping per CDNA5 ISA table
        AFrag af;
#pragma unroll
        for (int j = 0; j < 8; ++j) {
            int kk = k0 + ((j < 4) ? 0 : 16) + (j & 3) * 2 + g * 8;
            af.u[j] = *(const unsigned int*)(arow + kk);   // pair (K, K+1)
        }
        // B fragments from LDS: lane holds W column N=lane%16 of each tile, K mapping same as A
        const _Float16* blrow = ldsW + (size_t)k0 * LDS_STRIDE + (lane & 15);
        const _Float16* brrow = blrow + (size_t)K * LDS_STRIDE;
        BFrag bf[8];
#pragma unroll
        for (int n = 0; n < 8; ++n) {
#pragma unroll
            for (int s = 0; s < 16; ++s)
                bf[n].v[s] = blrow[(size_t)(((s < 8) ? 0 : 16) + g * 8 + (s & 7)) * LDS_STRIDE + n * 16];
        }
#pragma unroll
        for (int n = 0; n < 8; ++n)
            accL[n] = __builtin_amdgcn_wmma_f32_16x16x32_f16(
                false, af.v, false, bf[n].v, (short)0, accL[n], false, false);
#pragma unroll
        for (int n = 0; n < 8; ++n) {
#pragma unroll
            for (int s = 0; s < 16; ++s)
                bf[n].v[s] = brrow[(size_t)(((s < 8) ? 0 : 16) + g * 8 + (s & 7)) * LDS_STRIDE + n * 16];
        }
#pragma unroll
        for (int n = 0; n < 8; ++n)
            accR[n] = __builtin_amdgcn_wmma_f32_16x16x32_f16(
                false, af.v, false, bf[n].v, (short)0, accR[n], false, false);
    }
    // C/D layout: VGPR v -> M = v + 8*g ; N = lane%16
    int ncol = lane & 15;
#pragma unroll
    for (int n = 0; n < 8; ++n) {
        int c = n * 16 + ncol;
        float bL = biasL[c];
        float bR = biasR[c];
#pragma unroll
        for (int v = 0; v < 8; ++v) {
            size_t idx = (size_t)(row0 + v + 8 * g) * HIDF + c;
            YL[idx] = accL[n][v] + bL;
            YR[idx] = accR[n][v] + bR;
        }
    }
}

// ---------------- per-layer accumulator init with summed relation biases ----------------
__global__ __launch_bounds__(256)
void acc_init_kernel(float* __restrict__ acc, const float* __restrict__ biasR) {
    int t = blockIdx.x * 256 + threadIdx.x;
    if (t >= N_STOCKS * HIDF) return;
    int j = t & (HIDF - 1);
    float s = 0.f;
#pragma unroll
    for (int r = 0; r < R_REL; ++r) s += biasR[r * HIDF + j];
    acc[t] = s;
}

__global__ __launch_bounds__(256)
void maxsum_init_kernel(float* __restrict__ maxbuf, float* __restrict__ sumbuf) {
    int t = blockIdx.x * 256 + threadIdx.x;
    if (t >= N_STOCKS * HEADS) return;
    maxbuf[t] = -3.0e38f;
    sumbuf[t] = 0.f;
}

// ---------------- edge pass 1: score + atomic max per (dst, head) ----------------
__global__ __launch_bounds__(128)
void gat_score_kernel(const float* __restrict__ XL, const float* __restrict__ XR,
                      const int* __restrict__ esrc, const int* __restrict__ edst,
                      const float* __restrict__ att,       // [HEADS*CDIM]
                      float* __restrict__ escore, float* __restrict__ maxbuf) {
    int t = blockIdx.x * 128 + threadIdx.x;
    if (t >= EAUG * HEADS) return;
    int e = t >> 2, h = t & 3;
    int s, d;
    if (e < E_EDGES) { s = esrc[e]; d = edst[e]; }
    else             { s = d = e - E_EDGES; }
    const float* xl = XL + (size_t)s * HIDF + h * CDIM;
    const float* xr = XR + (size_t)d * HIDF + h * CDIM;
    const float* a  = att + h * CDIM;
    float sc = 0.f;
#pragma unroll
    for (int c = 0; c < CDIM; ++c) {
        float v = xl[c] + xr[c];
        v = (v > 0.f) ? v : 0.2f * v;           // leaky_relu 0.2
        sc += a[c] * v;
    }
    escore[(size_t)e * HEADS + h] = sc;
    __hip_atomic_fetch_max(&maxbuf[(size_t)d * HEADS + h], sc,
                           __ATOMIC_RELAXED, __HIP_MEMORY_SCOPE_AGENT);
}

// ---------------- edge pass 2: exp(score - max) + atomic sum ----------------
__global__ __launch_bounds__(128)
void gat_expsum_kernel(const int* __restrict__ edst,
                       float* __restrict__ escore, const float* __restrict__ maxbuf,
                       float* __restrict__ sumbuf) {
    int t = blockIdx.x * 128 + threadIdx.x;
    if (t >= EAUG * HEADS) return;
    int e = t >> 2, h = t & 3;
    int d = (e < E_EDGES) ? edst[e] : (e - E_EDGES);
    float m  = maxbuf[(size_t)d * HEADS + h];
    float ex = __expf(escore[(size_t)e * HEADS + h] - m);
    escore[(size_t)e * HEADS + h] = ex;
    atomicAdd(&sumbuf[(size_t)d * HEADS + h], ex);
}

// ---------------- edge pass 3: alpha-weighted scatter of XL[src] ----------------
__global__ __launch_bounds__(128)
void gat_scatter_kernel(const int* __restrict__ esrc, const int* __restrict__ edst,
                        const float* __restrict__ escore, const float* __restrict__ sumbuf,
                        const float* __restrict__ XL, float* __restrict__ acc) {
    int t = blockIdx.x * 128 + threadIdx.x;
    if (t >= EAUG * HEADS) return;
    int e = t >> 2, h = t & 3;
    int s, d;
    if (e < E_EDGES) { s = esrc[e]; d = edst[e]; }
    else             { s = d = e - E_EDGES; }
    float alpha = escore[(size_t)e * HEADS + h] /
                  (sumbuf[(size_t)d * HEADS + h] + 1e-16f);
    const float* xl = XL + (size_t)s * HIDF + h * CDIM;
    float*       ap = acc + (size_t)d * HIDF + h * CDIM;
#pragma unroll
    for (int c = 0; c < CDIM; ++c)
        atomicAdd(&ap[c], alpha * xl[c]);
}

// ---------------- ReLU (in place, f32) + re-quantize to f16 for next layer ----------------
__global__ __launch_bounds__(256)
void relu_cvt_kernel(float* __restrict__ acc, _Float16* __restrict__ h16) {
    int t = blockIdx.x * 256 + threadIdx.x;
    if (t >= N_STOCKS * HIDF) return;
    float v = acc[t];
    v = (v > 0.f) ? v : 0.f;
    acc[t] = v;
    h16[t] = (_Float16)v;
}

// ---------------- output MLP: relu(h@oW1+ob1)@oW2+ob2, one row per block ----------------
__global__ __launch_bounds__(64)
void mlp_kernel(const float* __restrict__ acc,
                const float* __restrict__ oW1, const float* __restrict__ ob1,
                const float* __restrict__ oW2, const float* __restrict__ ob2,
                float* __restrict__ out) {
    __shared__ float sh[64];
    int row = blockIdx.x, t = threadIdx.x;
    const float* hr = acc + (size_t)row * HIDF;
    float s = ob1[t];
#pragma unroll 8
    for (int k = 0; k < HIDF; ++k) s += hr[k] * oW1[k * 64 + t];
    s = (s > 0.f) ? s : 0.f;
    sh[t] = s * oW2[t];
    __syncthreads();
    for (int off = 32; off > 0; off >>= 1) {
        if (t < off) sh[t] += sh[t + off];
        __syncthreads();
    }
    if (t == 0) out[row] = sh[0] + ob2[0];
}

extern "C" void kernel_launch(void* const* d_in, const int* in_sizes, int n_in,
                              void* d_out, int out_size, void* d_ws, size_t ws_size,
                              hipStream_t stream) {
    const float* x       = (const float*)d_in[0];
    const int*   ei      = (const int*)  d_in[1];   // [R,2,E] int32
    const float* pearl_W = (const float*)d_in[2];
    const float* pearl_b = (const float*)d_in[3];
    const float* Wl[2]   = { (const float*)d_in[4],  (const float*)d_in[10] };
    const float* bl[2]   = { (const float*)d_in[5],  (const float*)d_in[11] };
    const float* Wr[2]   = { (const float*)d_in[6],  (const float*)d_in[12] };
    const float* br[2]   = { (const float*)d_in[7],  (const float*)d_in[13] };
    const float* att[2]  = { (const float*)d_in[8],  (const float*)d_in[14] };
    const float* biasL[2]= { (const float*)d_in[9],  (const float*)d_in[15] };
    const float* oW1 = (const float*)d_in[16];
    const float* ob1 = (const float*)d_in[17];
    const float* oW2 = (const float*)d_in[18];
    const float* ob2 = (const float*)d_in[19];
    float* out = (float*)d_out;

    // ---- workspace carve (256B aligned) ----
    char* wp = (char*)d_ws;
    auto take = [&](size_t bytes) -> void* {
        void* q = (void*)wp;
        wp += (bytes + 255) & ~(size_t)255;
        return q;
    };
    _Float16* h16  = (_Float16*)take((size_t)N_STOCKS * HIDF * 2);
    _Float16* WlH[2], *WrH[2];
    WlH[0] = (_Float16*)take((size_t)R_REL * TIN  * HIDF * 2);
    WrH[0] = (_Float16*)take((size_t)R_REL * TIN  * HIDF * 2);
    WlH[1] = (_Float16*)take((size_t)R_REL * HIDF * HIDF * 2);
    WrH[1] = (_Float16*)take((size_t)R_REL * HIDF * HIDF * 2);
    float* XL    = (float*)take((size_t)N_STOCKS * HIDF * 4);
    float* XR    = (float*)take((size_t)N_STOCKS * HIDF * 4);
    float* acc   = (float*)take((size_t)N_STOCKS * HIDF * 4);
    float* maxb  = (float*)take((size_t)N_STOCKS * HEADS * 4);
    float* sumb  = (float*)take((size_t)N_STOCKS * HEADS * 4);
    float* esc   = (float*)take((size_t)EAUG * HEADS * 4);

    // ---- convert weights to f16 once ----
    {
        int c0 = R_REL * TIN * HIDF, c1 = R_REL * HIDF * HIDF;
        cvt_f16_kernel<<<(c0 + 255) / 256, 256, 0, stream>>>(Wl[0], WlH[0], c0);
        cvt_f16_kernel<<<(c0 + 255) / 256, 256, 0, stream>>>(Wr[0], WrH[0], c0);
        cvt_f16_kernel<<<(c1 + 255) / 256, 256, 0, stream>>>(Wl[1], WlH[1], c1);
        cvt_f16_kernel<<<(c1 + 255) / 256, 256, 0, stream>>>(Wr[1], WrH[1], c1);
    }

    // ---- pearl PE + concat -> h16 ----
    pearl_concat_kernel<<<N_STOCKS, 32, 0, stream>>>(x, pearl_W, pearl_b, h16);

    const int rowTiles  = N_STOCKS / 16;                 // 1250 exactly
    const int gemmBlks  = (rowTiles + 3) / 4;
    const int edgeBlks  = (EAUG * HEADS + 127) / 128;
    const int nhBlks    = (N_STOCKS * HIDF + 255) / 256;
    const int msBlks    = (N_STOCKS * HEADS + 255) / 256;

    for (int layer = 0; layer < 2; ++layer) {
        const int K        = layer ? HIDF : TIN;
        const int wstride  = K * HIDF;
        const size_t ldsSz = (size_t)2 * K * LDS_STRIDE * sizeof(_Float16);

        acc_init_kernel<<<nhBlks, 256, 0, stream>>>(acc, biasL[layer]);

        for (int r = 0; r < R_REL; ++r) {
            const int* esrc = ei + ((size_t)r * 2 + 0) * E_EDGES;
            const int* edst = ei + ((size_t)r * 2 + 1) * E_EDGES;

            wmma_dual_lin_kernel<<<gemmBlks, 128, ldsSz, stream>>>(
                h16, K,
                WlH[layer] + (size_t)r * wstride, WrH[layer] + (size_t)r * wstride,
                bl[layer] + r * HIDF, br[layer] + r * HIDF,
                XL, XR, N_STOCKS);

            maxsum_init_kernel<<<msBlks, 256, 0, stream>>>(maxb, sumb);
            gat_score_kernel<<<edgeBlks, 128, 0, stream>>>(
                XL, XR, esrc, edst, att[layer] + r * HEADS * CDIM, esc, maxb);
            gat_expsum_kernel<<<edgeBlks, 128, 0, stream>>>(edst, esc, maxb, sumb);
            gat_scatter_kernel<<<edgeBlks, 128, 0, stream>>>(esrc, edst, esc, sumb, XL, acc);
        }
        // ReLU + produce f16 activations for next layer's GEMMs
        relu_cvt_kernel<<<nhBlks, 256, 0, stream>>>(acc, h16);
    }

    // ---- output MLP ----
    mlp_kernel<<<N_STOCKS, 64, 0, stream>>>(acc, oW1, ob1, oW2, ob2, out);
}
